// DepthCNN2d_4286377361688
// MI455X (gfx1250) — matrix operand order repaired and checked by reading, not compiled, not run on hardware
//
#include <hip/hip_runtime.h>

// DepthCNN2d on MI455X (gfx1250).
// GEMM runs on v_wmma_f32_16x16x32_f16 using a hi/lo _Float16 split of both
// operands (3 WMMAs per 32-K block: hh + hl + lh) -> fp32-grade accuracy at
// f16 matrix-pipe rates. Weights are pre-split/pre-swizzled ONCE into the
// per-lane WMMA A-operand order in d_ws (needs ws_size >= 294912 bytes).
//
// Shapes (from reference): C_IN=64, C_OUT=128, H=W=128, KK=576, B from sizes.

typedef __attribute__((ext_vector_type(16))) _Float16 v16h;
typedef __attribute__((ext_vector_type(8)))  _Float16 v8h;
typedef __attribute__((ext_vector_type(8)))  float    v8f;

#define C_IN    64
#define C_OUT   128
#define HDIM    128
#define WDIM    128
#define KKDEPTH 576                 // C_IN * 9
#define NKB     18                  // 576 / 32 K-blocks
#define TILE_P  64                  // pixels per workgroup (4 N-tiles / wave)
#define BSTR    584                 // LDS f16 stride per pixel (16B-aligned chunks)
#define AHI_ELEMS (C_OUT * KKDEPTH) // 73728 f16 per array

// ---------------------------------------------------------------------------
// Prep: split W2[o][kk] = Wflat[o*576+kk] into hi/lo f16, arranged in the
// A-operand order used here (contiguous k, same as B): index = (((mtile*18+kb)*32 + lane)*16 + i),
// lane half h holds K = kb*32 + 16h .. 16h+15 for row 16*mtile+(lane&15).
// ---------------------------------------------------------------------------
__global__ void depthcnn_prep_w(const float* __restrict__ w,
                                _Float16* __restrict__ ahi,
                                _Float16* __restrict__ alo)
{
    const int o    = blockIdx.x * 256 + threadIdx.x;   // 0..73727 (288 blocks x 256)
    const int i    = o & 15;
    const int lane = (o >> 4) & 31;
    const int wkb  = o >> 9;                           // mtile*18 + kb
    const int kb   = wkb % NKB;
    const int mt   = wkb / NKB;
    const int half = lane >> 4;
    const int row  = (mt << 4) + (lane & 15);
    const int kk   = (kb << 5) + (half << 4) + i;
    const float v  = w[row * KKDEPTH + kk];
    const _Float16 hi = (_Float16)v;
    const _Float16 lo = (_Float16)(v - (float)hi);     // exact residual in f32
    ahi[o] = hi;
    alo[o] = lo;
}

// ---------------------------------------------------------------------------
// Main kernel: one WG = 64 pixels (half an image row) x all 128 out channels.
// 8 waves; wave w owns M-tile (rows 16w..16w+15) and sweeps 4 N-tiles.
// ---------------------------------------------------------------------------
__launch_bounds__(256, 2)
__global__ void depthcnn_main(const float*    __restrict__ input,    // (B,64,128,128)
                              const float*    __restrict__ g_buffer, // (B,2,128,128)
                              const _Float16* __restrict__ ahi,      // arranged hi
                              const _Float16* __restrict__ alo,      // arranged lo
                              const float*    __restrict__ bias,     // (128)
                              float*          __restrict__ out)      // (B,128,128,128)
{
    extern __shared__ char smem[];
    _Float16* bhi  = (_Float16*)smem;                       // [TILE_P][BSTR]
    _Float16* blo  = bhi + TILE_P * BSTR;                   // [TILE_P][BSTR]
    float*    wf_s = (float*)(blo + TILE_P * BSTR);         // [9][TILE_P]

    const int blk = blockIdx.x;            // b*256 + y*2 + xhalf
    const int b   = blk >> 8;
    const int rem = blk & 255;
    const int y   = rem >> 1;
    const int x0  = (rem & 1) << 6;

    const int t = threadIdx.x;             // 0..255

    // ---- Phase 1a: bilateral weights wf[k][p] (one thread per pixel) ----
    if (t < TILE_P) {
        const int px = x0 + t;
        const float* gd = g_buffer + (((b << 1) + 1) << 14);   // depth plane
        const float dc  = 2.f * gd[(y << 7) + px] - 1.f;
        float e[9];
        float s = 0.f;
        #pragma unroll
        for (int i = 0; i < 3; ++i) {
            #pragma unroll
            for (int j = 0; j < 3; ++j) {
                const int yn = y + i - 1, xn = px + j - 1;
                const bool inb = (yn >= 0) & (yn < HDIM) & (xn >= 0) & (xn < WDIM);
                const float dn = inb ? (2.f * gd[(yn << 7) + xn] - 1.f) : 0.f;
                const float df = dn - dc;
                const float ev = __expf(-df * df);
                e[i * 3 + j] = ev;
                s += ev;
            }
        }
        const float inv = 9.f / s;
        #pragma unroll
        for (int k = 0; k < 9; ++k) wf_s[k * TILE_P + t] = e[k] * inv;
    }
    __syncthreads();

    // ---- Phase 1b: filt tile (576 x 64) -> hi/lo f16 in LDS ----
    {
        const int p  = t & 63;             // pixel
        const int cg = t >> 6;             // 0..3 -> 16 channels each
        const int x  = x0 + p;
        float w9[9];
        #pragma unroll
        for (int k = 0; k < 9; ++k) w9[k] = wf_s[k * TILE_P + p];

        #pragma unroll 4
        for (int cc = 0; cc < 16; ++cc) {
            const int c = (cg << 4) + cc;
            const float* inc = input + (((b << 6) + c) << 14);
            const int base = p * BSTR + c * 9;
            #pragma unroll
            for (int i = 0; i < 3; ++i) {
                #pragma unroll
                for (int j = 0; j < 3; ++j) {
                    const int yn = y + i - 1, xn = x + j - 1;
                    const bool inb = (yn >= 0) & (yn < HDIM) & (xn >= 0) & (xn < WDIM);
                    const float v  = (inb ? inc[(yn << 7) + xn] : 0.f) * w9[i * 3 + j];
                    const _Float16 hi = (_Float16)v;
                    const _Float16 lo = (_Float16)(v - (float)hi);
                    bhi[base + i * 3 + j] = hi;
                    blo[base + i * 3 + j] = lo;
                }
            }
        }
    }
    __syncthreads();

    // ---- Phase 2: f16x2-split WMMA GEMM ----
    // A (16x32 f16) per lane: 16 contiguous f16 from the arranged array.
    // B (32x16 f16) per lane: N = lane&15, K = kb*32 + 16*half .. +15 contiguous.
    const int lane = t & 31;
    const int wave = t >> 5;               // M-tile
    const int half = lane >> 4;
    const int nn   = lane & 15;

    v8f acc0 = {}, acc1 = {}, acc2 = {}, acc3 = {};

    #pragma unroll 3
    for (int kb = 0; kb < NKB; ++kb) {
        const int aoff = (((wave * NKB + kb) << 5) + lane) << 4;   // *16 f16
        const v16h ah = *(const v16h*)(ahi + aoff);                // 32B coalesced
        const v16h al = *(const v16h*)(alo + aoff);

        const int bko = (kb << 5) + (half << 4);
        #pragma unroll
        for (int nt = 0; nt < 4; ++nt) {
            const _Float16* bp = bhi + ((nt << 4) + nn) * BSTR + bko;
            const v8h h0 = *(const v8h*)(bp);
            const v8h h1 = *(const v8h*)(bp + 8);
            const v8h l0 = *(const v8h*)(bp + TILE_P * BSTR);
            const v8h l1 = *(const v8h*)(bp + TILE_P * BSTR + 8);
            const v16h bh = __builtin_shufflevector(h0, h1, 0,1,2,3,4,5,6,7,8,9,10,11,12,13,14,15);
            const v16h bl = __builtin_shufflevector(l0, l1, 0,1,2,3,4,5,6,7,8,9,10,11,12,13,14,15);

            v8f a = (nt == 0) ? acc0 : (nt == 1) ? acc1 : (nt == 2) ? acc2 : acc3;
            a = __builtin_amdgcn_wmma_f32_16x16x32_f16(false, ah, false, bh, (short)0, a, false, false);
            a = __builtin_amdgcn_wmma_f32_16x16x32_f16(false, ah, false, bl, (short)0, a, false, false);
            a = __builtin_amdgcn_wmma_f32_16x16x32_f16(false, al, false, bh, (short)0, a, false, false);
            if      (nt == 0) acc0 = a;
            else if (nt == 1) acc1 = a;
            else if (nt == 2) acc2 = a;
            else              acc3 = a;
        }
    }

    // ---- Epilogue: VGPR j holds (M = j + 8*half, N = nn) ----
    float* outb = out + ((b * C_OUT) << 14);
    #pragma unroll
    for (int nt = 0; nt < 4; ++nt) {
        const v8f a = (nt == 0) ? acc0 : (nt == 1) ? acc1 : (nt == 2) ? acc2 : acc3;
        const int pix = (y << 7) + x0 + (nt << 4) + nn;
        #pragma unroll
        for (int j = 0; j < 8; ++j) {
            const int o = (wave << 4) + (half << 3) + j;
            outb[(o << 14) + pix] = a[j] + bias[o];
        }
    }
}

extern "C" void kernel_launch(void* const* d_in, const int* in_sizes, int n_in,
                              void* d_out, int out_size, void* d_ws, size_t ws_size,
                              hipStream_t stream) {
    const float* input   = (const float*)d_in[0];
    const float* gbuf    = (const float*)d_in[1];
    const float* weights = (const float*)d_in[2];
    const float* bias    = (const float*)d_in[3];
    float* out = (float*)d_out;

    _Float16* ahi = (_Float16*)d_ws;                 // 147456 B
    _Float16* alo = ahi + AHI_ELEMS;                 // 147456 B  (ws >= 294912 B)

    // Split + swizzle weights once (deterministic: same inputs -> same ws).
    depthcnn_prep_w<<<dim3(AHI_ELEMS / 256), dim3(256), 0, stream>>>(weights, ahi, alo);

    const int B = in_sizes[0] / (C_IN * HDIM * WDIM);            // 8
    const int nblocks = B * HDIM * (WDIM / TILE_P);              // 2048

    const size_t shmem = (size_t)(2 * TILE_P * BSTR) * sizeof(_Float16)  // bhi+blo
                       + (size_t)(9 * TILE_P) * sizeof(float);           // wf
    static_assert(2 * TILE_P * BSTR * sizeof(_Float16) + 9 * TILE_P * sizeof(float)
                  < 320 * 1024, "LDS over WGP budget");
    hipFuncSetAttribute(reinterpret_cast<const void*>(depthcnn_main),
                        hipFuncAttributeMaxDynamicSharedMemorySize, (int)shmem);

    depthcnn_main<<<dim3(nblocks), dim3(256), shmem, stream>>>(
        input, gbuf, ahi, alo, bias, out);
}
